// PerceiverAttention_6708738916614
// MI455X (gfx1250) — matrix operand order repaired and checked by reading, not compiled, not run on hardware
//
#include <hip/hip_runtime.h>
#include <math.h>

// ---------------------------------------------------------------------------
// PerceiverAttention for MI455X (gfx1250): bf16 WMMA pipeline.
// B=8, M=4096, N=256, DIM=1024, HEADS=16, DH=64, J = M+N = 4352.
// GEMMs register-blocked 32x64 per wave; all fragment loads issued before the
// WMMA burst so the scheduler can use partial s_wait_loadcnt overlap.
// ---------------------------------------------------------------------------

typedef __bf16 bf16_t;
typedef __attribute__((ext_vector_type(16))) __bf16        v16bf;
typedef __attribute__((ext_vector_type(8)))  float         v8f;
typedef __attribute__((ext_vector_type(4)))  unsigned int  u32x4;

#define DIMQ    1024
#define HEADS   16
#define DH      64
#define INNERQ  1024
#define BBATCH  8
#define MROWS   4096
#define NLAT    256
#define JTOT    4352

union FragU { u32x4 q[2]; v16bf v; };
union Pack8 { bf16_t h[8]; u32x4 u; };

// A-fragment: 16x32 bf16, row-major source (stride lda elements).
// Lane (hf = lane>>4, r = lane&15) holds row r; element e -> k = (e>>3)*16 + hf*8 + (e&7).
static __device__ __forceinline__ v16bf load_a_frag(const bf16_t* base, int lda, int lane) {
    int hf = lane >> 4, r = lane & 15;
    const bf16_t* p = base + (size_t)r * lda + hf * 8;
    FragU f;
    f.q[0] = *(const u32x4*)(p);
    f.q[1] = *(const u32x4*)(p + 16);
    return f.v;
}

// B-fragment: 32x16 bf16 from BT (N x K row-major, stride ldbt).
// Lane holds column n = lane&15; element e -> k = (e>>3)*16 + (lane>>4)*8 + (e&7).
static __device__ __forceinline__ v16bf load_b_frag(const bf16_t* bt, int ldbt, int lane) {
    int hf = lane >> 4, n = lane & 15;
    const bf16_t* p = bt + (size_t)n * ldbt + hf * 8;
    FragU f;
    f.q[0] = *(const u32x4*)(p);
    f.q[1] = *(const u32x4*)(p + 16);
    return f.v;
}

static __device__ __forceinline__ v8f wmma_bf16(v16bf a, v16bf b, v8f c) {
    return __builtin_amdgcn_wmma_f32_16x16x32_bf16(false, a, false, b, (short)0, c, false, false);
}

// ---------------------------------------------------------------------------
// Weight cast + transpose: W (K x N f32, row-major) -> WT (N x K bf16, row-major)
// ---------------------------------------------------------------------------
__global__ void wcastT_kernel(const float* __restrict__ W, bf16_t* __restrict__ WT,
                              int K, int N) {
    size_t idx = (size_t)blockIdx.x * blockDim.x + threadIdx.x;
    if (idx >= (size_t)K * N) return;
    int k = (int)(idx / N);
    int n = (int)(idx % N);
    WT[(size_t)n * K + k] = (bf16_t)W[idx];
}

// ---------------------------------------------------------------------------
// LayerNorm f32 row (1024) -> bf16 row into kv_in[b][dst_off + rr][1024]
// ---------------------------------------------------------------------------
__global__ void ln_cast_kernel(const float* __restrict__ src,
                               const float* __restrict__ g,
                               const float* __restrict__ bvec,
                               bf16_t* __restrict__ dst,
                               int rows_per_batch, int dst_off) {
    __shared__ float s1[256];
    __shared__ float s2[256];
    int row   = blockIdx.x;
    int batch = row / rows_per_batch;
    int rr    = row % rows_per_batch;
    const float* s = src + (size_t)row * DIMQ;
    bf16_t*      d = dst + ((size_t)batch * JTOT + dst_off + rr) * DIMQ;

    int t = threadIdx.x;
    float4 v = *(const float4*)(s + t * 4);
    float sum = v.x + v.y + v.z + v.w;
    float sq  = v.x * v.x + v.y * v.y + v.z * v.z + v.w * v.w;
    s1[t] = sum; s2[t] = sq;
    __syncthreads();
    for (int st = 128; st > 0; st >>= 1) {
        if (t < st) { s1[t] += s1[t + st]; s2[t] += s2[t + st]; }
        __syncthreads();
    }
    float mu   = s1[0] * (1.0f / DIMQ);
    float var  = s2[0] * (1.0f / DIMQ) - mu * mu;
    float rstd = rsqrtf(var + 1e-5f);

    float vals[4] = {v.x, v.y, v.z, v.w};
#pragma unroll
    for (int c = 0; c < 4; ++c) {
        int idx = t * 4 + c;
        d[idx] = (bf16_t)((vals[c] - mu) * rstd * g[idx] + bvec[idx]);
    }
}

// ---------------------------------------------------------------------------
// Generic 32x64-per-wave blocked accumulate: A rows m0..m0+31, BT rows f0..f0+63.
// All 12 b128 loads of a K-step are issued before the 8-WMMA burst.
// ---------------------------------------------------------------------------
#define GEMM_BODY(Aptr, BTptr, K)                                            \
    v8f acc[2][4];                                                           \
    _Pragma("unroll") for (int mi = 0; mi < 2; ++mi)                         \
        _Pragma("unroll") for (int nb = 0; nb < 4; ++nb)                     \
            acc[mi][nb] = (v8f){0.f,0.f,0.f,0.f,0.f,0.f,0.f,0.f};            \
    for (int k0 = 0; k0 < (K); k0 += 32) {                                   \
        v16bf a0 = load_a_frag((Aptr) + k0, (K), lane);                      \
        v16bf a1 = load_a_frag((Aptr) + (size_t)16 * (K) + k0, (K), lane);   \
        v16bf b0 = load_b_frag((BTptr) + (size_t)0  * (K) + k0, (K), lane);  \
        v16bf b1 = load_b_frag((BTptr) + (size_t)16 * (K) + k0, (K), lane);  \
        v16bf b2 = load_b_frag((BTptr) + (size_t)32 * (K) + k0, (K), lane);  \
        v16bf b3 = load_b_frag((BTptr) + (size_t)48 * (K) + k0, (K), lane);  \
        acc[0][0] = wmma_bf16(a0, b0, acc[0][0]);                            \
        acc[1][0] = wmma_bf16(a1, b0, acc[1][0]);                            \
        acc[0][1] = wmma_bf16(a0, b1, acc[0][1]);                            \
        acc[1][1] = wmma_bf16(a1, b1, acc[1][1]);                            \
        acc[0][2] = wmma_bf16(a0, b2, acc[0][2]);                            \
        acc[1][2] = wmma_bf16(a1, b2, acc[1][2]);                            \
        acc[0][3] = wmma_bf16(a0, b3, acc[0][3]);                            \
        acc[1][3] = wmma_bf16(a1, b3, acc[1][3]);                            \
    }

// ---------------------------------------------------------------------------
// q = ln @ Wq, scaled by dh^-0.5, stored bf16 [b][256][1024]
// ---------------------------------------------------------------------------
__global__ void gemm_q_kernel(const bf16_t* __restrict__ kv_in,
                              const bf16_t* __restrict__ WqT,
                              bf16_t* __restrict__ qbuf) {
    int lane = threadIdx.x & 31, w = threadIdx.x >> 5;
    int b  = blockIdx.z;
    int m0 = blockIdx.x * 32;
    int f0 = (blockIdx.y * 8 + w) * 64;
    const bf16_t* A  = kv_in + ((size_t)b * JTOT + MROWS + m0) * DIMQ;
    const bf16_t* BT = WqT + (size_t)f0 * DIMQ;
    GEMM_BODY(A, BT, DIMQ)
    int hf = lane >> 4, nn = lane & 15;
    bf16_t* out = qbuf + (size_t)b * NLAT * INNERQ;
#pragma unroll
    for (int mi = 0; mi < 2; ++mi)
#pragma unroll
        for (int i = 0; i < 8; ++i) {
            size_t rowoff = (size_t)(m0 + mi * 16 + i + 8 * hf) * INNERQ + f0 + nn;
#pragma unroll
            for (int nb = 0; nb < 4; ++nb)
                out[rowoff + nb * 16] = (bf16_t)(acc[mi][nb][i] * 0.125f); // 64^-0.5
        }
}

// ---------------------------------------------------------------------------
// kv = kv_in @ Wkv ; K part -> kbuf [b][h][j][d] natural, V part -> vT [b][h][d][j]
// The 64-wide N tile equals exactly one head's d-range.
// ---------------------------------------------------------------------------
__global__ void gemm_kv_kernel(const bf16_t* __restrict__ kv_in,
                               const bf16_t* __restrict__ WkvT,
                               bf16_t* __restrict__ kbuf,
                               bf16_t* __restrict__ vTbuf) {
    int lane = threadIdx.x & 31, w = threadIdx.x >> 5;
    int b  = blockIdx.z;
    int j0 = blockIdx.x * 32;
    int f0 = (blockIdx.y * 8 + w) * 64;
    const bf16_t* A  = kv_in + ((size_t)b * JTOT + j0) * DIMQ;
    const bf16_t* BT = WkvT + (size_t)f0 * DIMQ;
    GEMM_BODY(A, BT, DIMQ)
    int hf = lane >> 4, nn = lane & 15;
    if (f0 < INNERQ) {                 // K features: [b][h][j][d]
        int h = f0 / DH;
        bf16_t* kb = kbuf + (size_t)(b * HEADS + h) * JTOT * DH;
#pragma unroll
        for (int mi = 0; mi < 2; ++mi)
#pragma unroll
            for (int i = 0; i < 8; ++i) {
                size_t rowoff = (size_t)(j0 + mi * 16 + i + 8 * hf) * DH + nn;
#pragma unroll
                for (int nb = 0; nb < 4; ++nb)
                    kb[rowoff + nb * 16] = (bf16_t)acc[mi][nb][i];
            }
    } else {                           // V features: transposed [b][h][d][j]
        int h = (f0 - INNERQ) / DH;
        bf16_t* vt = vTbuf + (size_t)(b * HEADS + h) * DH * JTOT;
#pragma unroll
        for (int nb = 0; nb < 4; ++nb) {
            int d = nb * 16 + nn;
#pragma unroll
            for (int mi = 0; mi < 2; ++mi) {
                Pack8 pk;
#pragma unroll
                for (int i = 0; i < 8; ++i) pk.h[i] = (bf16_t)acc[mi][nb][i];
                *(u32x4*)(vt + (size_t)d * JTOT + j0 + mi * 16 + 8 * hf) = pk.u;
            }
        }
    }
}

// ---------------------------------------------------------------------------
// Flash attention per (b, h, 16-query tile). One wave per tile, 4 waves/block.
// ---------------------------------------------------------------------------
__global__ void attn_kernel(const bf16_t* __restrict__ qbuf,
                            const bf16_t* __restrict__ kbuf,
                            const bf16_t* __restrict__ vTbuf,
                            const int*    __restrict__ xmask,
                            bf16_t* __restrict__ attn_out) {
    __shared__ __align__(16) bf16_t pbuf[4][16][32];
    int lane = threadIdx.x & 31, w = threadIdx.x >> 5;
    int hf = lane >> 4, nn = lane & 15;
    int b = blockIdx.z, h = blockIdx.y;
    int q0 = (blockIdx.x * 4 + w) * 16;
    size_t bh = (size_t)(b * HEADS + h);

    const bf16_t* qbase = qbuf + ((size_t)b * NLAT + q0) * INNERQ + h * DH;
    v16bf qa0 = load_a_frag(qbase,      INNERQ, lane);
    v16bf qa1 = load_a_frag(qbase + 32, INNERQ, lane);

    const bf16_t* kb   = kbuf  + bh * (size_t)JTOT * DH;
    const bf16_t* vt   = vTbuf + bh * (size_t)DH * JTOT;
    const int*    mrow = xmask + b * MROWS;

    v8f o0 = {0.f,0.f,0.f,0.f,0.f,0.f,0.f,0.f};
    v8f o1 = o0, o2 = o0, o3 = o0;
    float rm[8], rl[8];
#pragma unroll
    for (int i = 0; i < 8; ++i) { rm[i] = -__builtin_inff(); rl[i] = 0.f; }

    for (int j0 = 0; j0 < JTOT; j0 += 32) {
        // ---- sim tiles: preload 4 K-fragments, then 4 WMMAs -------------
        v16bf kb00 = load_b_frag(kb + (size_t)j0 * DH,             DH, lane);
        v16bf kb01 = load_b_frag(kb + (size_t)j0 * DH + 32,        DH, lane);
        v16bf kb10 = load_b_frag(kb + (size_t)(j0 + 16) * DH,      DH, lane);
        v16bf kb11 = load_b_frag(kb + (size_t)(j0 + 16) * DH + 32, DH, lane);
        v8f s0 = {0.f,0.f,0.f,0.f,0.f,0.f,0.f,0.f};
        v8f s1 = s0;
        s0 = wmma_bf16(qa0, kb00, s0);
        s0 = wmma_bf16(qa1, kb01, s0);
        s1 = wmma_bf16(qa0, kb10, s1);
        s1 = wmma_bf16(qa1, kb11, s1);

        int jj0 = j0 + nn, jj1 = j0 + 16 + nn;
        bool msk0 = (jj0 < MROWS) && (mrow[jj0] != 0);
        bool msk1 = (jj1 < MROWS) && (mrow[jj1] != 0);

#pragma unroll
        for (int i = 0; i < 8; ++i) {
            float a0 = msk0 ? -__builtin_inff() : s0[i];
            float a1 = msk1 ? -__builtin_inff() : s1[i];
            float bm = fmaxf(a0, a1);
#pragma unroll
            for (int dlt = 8; dlt >= 1; dlt >>= 1)
                bm = fmaxf(bm, __shfl_xor(bm, dlt, 32));
            float nm    = fmaxf(rm[i], bm);
            float alpha = (nm == -__builtin_inff()) ? 1.0f : __expf(rm[i] - nm);
            float p0 = msk0 ? 0.f : __expf(a0 - nm);
            float p1 = msk1 ? 0.f : __expf(a1 - nm);
            float ps = p0 + p1;
#pragma unroll
            for (int dlt = 8; dlt >= 1; dlt >>= 1)
                ps += __shfl_xor(ps, dlt, 32);
            rl[i] = rl[i] * alpha + ps;
            rm[i] = nm;
            o0[i] *= alpha; o1[i] *= alpha; o2[i] *= alpha; o3[i] *= alpha;
            int r = i + 8 * hf;
            pbuf[w][r][nn]      = (bf16_t)p0;
            pbuf[w][r][nn + 16] = (bf16_t)p1;
        }
        // ---- P@V: preload 4 V-fragments + LDS A-frag, then 4 WMMAs ------
        v16bf vb0 = load_b_frag(vt + j0,                     JTOT, lane);
        v16bf vb1 = load_b_frag(vt + (size_t)16 * JTOT + j0, JTOT, lane);
        v16bf vb2 = load_b_frag(vt + (size_t)32 * JTOT + j0, JTOT, lane);
        v16bf vb3 = load_b_frag(vt + (size_t)48 * JTOT + j0, JTOT, lane);
        v16bf pa  = load_a_frag(&pbuf[w][0][0], 32, lane);  // C-frag -> A-frag reshape
        o0 = wmma_bf16(pa, vb0, o0);
        o1 = wmma_bf16(pa, vb1, o1);
        o2 = wmma_bf16(pa, vb2, o2);
        o3 = wmma_bf16(pa, vb3, o3);
    }

    bf16_t* ob = attn_out + ((size_t)b * NLAT + q0) * INNERQ + h * DH;
#pragma unroll
    for (int i = 0; i < 8; ++i) {
        float inv = (rl[i] > 0.f) ? (1.0f / rl[i]) : 0.f;
        size_t rowoff = (size_t)(i + 8 * hf) * INNERQ;
        ob[rowoff +      nn] = (bf16_t)(o0[i] * inv);
        ob[rowoff + 16 + nn] = (bf16_t)(o1[i] * inv);
        ob[rowoff + 32 + nn] = (bf16_t)(o2[i] * inv);
        ob[rowoff + 48 + nn] = (bf16_t)(o3[i] * inv);
    }
}

// ---------------------------------------------------------------------------
// final = attn_out @ Wo -> f32 d_out [8*256][1024]
// ---------------------------------------------------------------------------
__global__ void gemm_out_kernel(const bf16_t* __restrict__ attn,
                                const bf16_t* __restrict__ WoT,
                                float* __restrict__ out) {
    int lane = threadIdx.x & 31, w = threadIdx.x >> 5;
    int m0 = blockIdx.x * 32;
    int f0 = (blockIdx.y * 8 + w) * 64;
    const bf16_t* A  = attn + (size_t)m0 * INNERQ;
    const bf16_t* BT = WoT + (size_t)f0 * INNERQ;
    GEMM_BODY(A, BT, INNERQ)
    int hf = lane >> 4, nn = lane & 15;
#pragma unroll
    for (int mi = 0; mi < 2; ++mi)
#pragma unroll
        for (int i = 0; i < 8; ++i) {
            size_t rowoff = (size_t)(m0 + mi * 16 + i + 8 * hf) * DIMQ + f0 + nn;
#pragma unroll
            for (int nb = 0; nb < 4; ++nb)
                out[rowoff + nb * 16] = acc[mi][nb][i];
        }
}

// ---------------------------------------------------------------------------
extern "C" void kernel_launch(void* const* d_in, const int* in_sizes, int n_in,
                              void* d_out, int out_size, void* d_ws, size_t ws_size,
                              hipStream_t stream) {
    const float* x       = (const float*)d_in[0];
    const float* latents = (const float*)d_in[1];
    const int*   xmask   = (const int*)  d_in[2];
    const float* nm_g    = (const float*)d_in[3];
    const float* nm_b    = (const float*)d_in[4];
    const float* nl_g    = (const float*)d_in[5];
    const float* nl_b    = (const float*)d_in[6];
    const float* Wq      = (const float*)d_in[7];
    const float* Wkv     = (const float*)d_in[8];
    const float* Wo      = (const float*)d_in[9];
    float* out = (float*)d_out;

    char* ws = (char*)d_ws;
    size_t off = 0;
    auto alloc = [&](size_t bytes) -> char* {
        char* p = ws + off;
        off += (bytes + 255) & ~(size_t)255;
        return p;
    };
    bf16_t* kv_in = (bf16_t*)alloc((size_t)BBATCH * JTOT * DIMQ * 2);   // LN(x)||LN(lat)
    bf16_t* qbuf  = (bf16_t*)alloc((size_t)BBATCH * NLAT * INNERQ * 2);
    bf16_t* kbuf  = (bf16_t*)alloc((size_t)BBATCH * HEADS * JTOT * DH * 2);
    bf16_t* vTbuf = (bf16_t*)alloc((size_t)BBATCH * HEADS * DH * JTOT * 2);
    bf16_t* attn  = (bf16_t*)alloc((size_t)BBATCH * NLAT * INNERQ * 2);
    bf16_t* WqT   = (bf16_t*)alloc((size_t)DIMQ * INNERQ * 2);
    bf16_t* WkvT  = (bf16_t*)alloc((size_t)DIMQ * 2 * INNERQ * 2);
    bf16_t* WoT   = (bf16_t*)alloc((size_t)INNERQ * DIMQ * 2);

    // 1) weights -> bf16 transposed
    wcastT_kernel<<<(DIMQ * INNERQ) / 256,     256, 0, stream>>>(Wq,  WqT,  DIMQ,   INNERQ);
    wcastT_kernel<<<(DIMQ * 2 * INNERQ) / 256, 256, 0, stream>>>(Wkv, WkvT, DIMQ,   2 * INNERQ);
    wcastT_kernel<<<(INNERQ * DIMQ) / 256,     256, 0, stream>>>(Wo,  WoT,  INNERQ, DIMQ);

    // 2) layernorm + cast into concatenated kv_in
    ln_cast_kernel<<<BBATCH * MROWS, 256, 0, stream>>>(x,       nm_g, nm_b, kv_in, MROWS, 0);
    ln_cast_kernel<<<BBATCH * NLAT,  256, 0, stream>>>(latents, nl_g, nl_b, kv_in, NLAT,  MROWS);

    // 3) projections (WMMA GEMMs, 32x64 per wave)
    gemm_q_kernel <<<dim3(NLAT / 32, INNERQ / 512, BBATCH),       256, 0, stream>>>(kv_in, WqT,  qbuf);
    gemm_kv_kernel<<<dim3(JTOT / 32, (2 * INNERQ) / 512, BBATCH), 256, 0, stream>>>(kv_in, WkvT, kbuf, vTbuf);

    // 4) flash attention (WMMA)
    attn_kernel<<<dim3((NLAT / 16) / 4, HEADS, BBATCH), 128, 0, stream>>>(qbuf, kbuf, vTbuf, xmask, attn);

    // 5) output projection -> f32
    gemm_out_kernel<<<dim3((BBATCH * NLAT) / 32, DIMQ / 512, 1), 256, 0, stream>>>(attn, WoT, out);
}
